// SelfAttention_49160195670600
// MI455X (gfx1250) — matrix operand order repaired and checked by reading, not compiled, not run on hardware
//
#include <hip/hip_runtime.h>
#include <hip/hip_bf16.h>

// MI455X (gfx1250) flash-style self-attention with column (query-axis) softmax.
//   B=8, L=2048, H=512.  All matmuls via v_wmma_f32_16x16x32_bf16, f32 accum.
// Pass 0: x f32 -> xb (row-major bf16) + xt (transposed bf16), LDS-tiled so
//         both global reads and writes are coalesced.
// Pass 1: per-k-column logit max m[b,k] and sum s[b,k] (online softmax over q).
//         k-tile (16x512) staged in LDS once via GLOBAL_LOAD_ASYNC_TO_LDS_B128.
// Pass 2: recompute score tiles, P = exp(logit - m[k]) / s[k], att = P @ X.
//         q-tile (16x512) staged in LDS once via async copy.

#define B_ 8
#define L_ 2048
#define H_ 512
#define INV_SQRT_DK 0.044194173824159216f /* 1/sqrt(512) */
#define XPAD 520  /* 512 + 8 bf16 row pad: 1040 B row stride -> bank-staggered */

typedef __bf16 bf16;
typedef __attribute__((ext_vector_type(16))) __bf16 v16bf;
typedef __attribute__((ext_vector_type(2)))  __bf16 bf16x2;
typedef __attribute__((ext_vector_type(8)))  float  v8f;

// ---- WMMA operand loaders (CDNA5 16-bit layouts, ISA 7.12.2) -------------
// A matrix 16x32: lane m = L&15, half = L>>4; VGPR v holds K pair starting at
//   kb = (v&3)*2 + half*8 + (v>>2)*16.  `row` points at A[m][0] of this K-slice.
static __device__ inline v16bf load_opA(const bf16* row, int half) {
  v16bf a;
#pragma unroll
  for (int v = 0; v < 8; ++v) {
    const int kb = ((v & 3) << 1) + (half << 3) + ((v >> 2) << 4);
    bf16x2 p = *reinterpret_cast<const bf16x2*>(row + kb);
    a[2 * v]     = p.x;
    a[2 * v + 1] = p.y;
  }
  return a;
}

// B matrix 32x16: lane n = L&15, half = L>>4; VGPR v holds K pair starting at
//   kb = (v&3)*2 + half*8 + (v>>2)*16.  `krun` points at B[0][n] with the K axis contiguous.
static __device__ inline v16bf load_opB(const bf16* krun, int half) {
  v16bf b;
#pragma unroll
  for (int v = 0; v < 8; ++v) {
    const int kb = ((v & 3) << 1) + (half << 3) + ((v >> 2) << 4);
    bf16x2 p = *reinterpret_cast<const bf16x2*>(krun + kb);
    b[2 * v]     = p.x;
    b[2 * v + 1] = p.y;
  }
  return b;
}

// ---- Async global->LDS staging of a 16x512 bf16 tile (16 KB) -------------
// 1024 chunks of 16 B; 128 threads x 8 chunks.  Source rows are contiguous
// (full xb rows); LDS rows are padded to XPAD bf16.  Generic pointers to LDS
// carry the LDS byte offset in their low 32 bits (aperture rule, ISA 10.2).
static __device__ inline void stage_tile_async(const bf16* gsrc, bf16 (*dst)[XPAD],
                                               int tid) {
  const unsigned lds0 = (unsigned)(size_t)&dst[0][0];
#pragma unroll
  for (int i = 0; i < 8; ++i) {
    const int cid = tid + 128 * i;          // chunk id 0..1023
    const int m   = cid >> 6;               // row (64 chunks of 16 B per row)
    const int c   = (cid & 63) << 4;        // byte offset within row
    const unsigned lds_off = lds0 + (unsigned)(m * (XPAD * 2) + c);
    const unsigned long long ga =
        (unsigned long long)(size_t)((const char*)gsrc + m * (H_ * 2) + c);
    asm volatile("global_load_async_to_lds_b128 %0, %1, off"
                 :: "v"(lds_off), "v"(ga) : "memory");
  }
  asm volatile("s_wait_asynccnt 0x0" ::: "memory");
}

// ---- Pass 0: LDS-tiled convert/transpose, fully coalesced ----------------
// 64(l) x 64(d) tile per 256-thread block: float2 reads, bf16x2 writes for
// both the row-major copy (xb) and the transposed copy (xt).
__global__ void k_convert(const float* __restrict__ x,
                          bf16* __restrict__ xb, bf16* __restrict__ xt) {
  __shared__ __attribute__((aligned(16))) bf16 Xs[64][66];  // odd DW stride
  const int t  = threadIdx.x;
  const int nd = H_ / 64;                 // 8
  const int nl = L_ / 64;                 // 32
  const int b  = blockIdx.x / (nd * nl);
  const int r2 = blockIdx.x % (nd * nl);
  const int l0 = (r2 / nd) * 64;
  const int d0 = (r2 % nd) * 64;

#pragma unroll
  for (int i = 0; i < 8; ++i) {
    const int p = t + 256 * i;            // pair id 0..2047
    const int r = p >> 5;                 // tile row (l)
    const int c = (p & 31) << 1;          // tile col (d), pair base
    const long goff = ((long)b * L_ + l0 + r) * H_ + d0 + c;
    const float2 v = *reinterpret_cast<const float2*>(x + goff);
    bf16x2 h;
    h.x = (bf16)v.x;
    h.y = (bf16)v.y;
    *reinterpret_cast<bf16x2*>(xb + goff) = h;
    Xs[r][c]     = h.x;
    Xs[r][c + 1] = h.y;
  }
  __syncthreads();
#pragma unroll
  for (int i = 0; i < 8; ++i) {
    const int p  = t + 256 * i;
    const int dd = p >> 5;                // xt tile row (d)
    const int lp = (p & 31) << 1;         // xt tile col (l), pair base
    bf16x2 h;
    h.x = Xs[lp][dd];
    h.y = Xs[lp + 1][dd];
    *reinterpret_cast<bf16x2*>(xt + ((long)b * H_ + d0 + dd) * L_ + l0 + lp) = h;
  }
}

// ---- Pass 1: per-column (k) logit max & exp-sum over the q axis ----------
__global__ void k_colstats(const bf16* __restrict__ xb,
                           const unsigned char* __restrict__ mask,
                           float* __restrict__ Mcol, float* __restrict__ Scol) {
  __shared__ __attribute__((aligned(16))) bf16 Xk[16][XPAD];  // k-tile, reused 32x
  __shared__ float red_m[4][32];
  __shared__ float red_s[4][32];
  const int kt   = blockIdx.x % (L_ / 16);
  const int b    = blockIdx.x / (L_ / 16);
  const int k0   = kt * 16;
  const int wave = threadIdx.x >> 5;
  const int lane = threadIdx.x & 31;
  const int n    = lane & 15;
  const int half = lane >> 4;

  const bf16* xbase = xb + (long)b * L_ * H_;

  // stage rows k0..k0+15 into LDS (async), then make visible to all waves
  stage_tile_async(xbase + (long)k0 * H_, Xk, threadIdx.x);
  __syncthreads();

  float mx = -3.0e38f;
  float sm = 0.0f;

  for (int qt = wave; qt < L_ / 16; qt += 4) {
    const int q0 = qt * 16;
    const bf16* aRow = xbase + (long)(q0 + n) * H_;  // m = lane&15 == n
    // prefetch the 8 mask bytes so their latency hides under the WMMAs
    const unsigned char* mrow = mask + ((long)b * L_ + q0) * L_ + (k0 + n);
    unsigned char mv[8];
#pragma unroll
    for (int j = 0; j < 8; ++j) mv[j] = mrow[(long)(j + 8 * half) * L_];

    v8f acc = {};
#pragma unroll 4
    for (int d0 = 0; d0 < H_; d0 += 32) {
      v16bf a  = load_opA(aRow + d0, half);
      v16bf bm = load_opB(&Xk[n][d0], half);   // ds_load_b128 x2
      acc = __builtin_amdgcn_wmma_f32_16x16x32_bf16(false, a, false, bm,
                                                    (short)0, acc, false, false);
    }
    // tile-local stats (shallow max tree, independent exps), then one merge
    float lg[8];
#pragma unroll
    for (int j = 0; j < 8; ++j)
      lg[j] = (mv[j] ? -1.0e9f : acc[j]) * INV_SQRT_DK;
    const float ml = fmaxf(fmaxf(fmaxf(lg[0], lg[1]), fmaxf(lg[2], lg[3])),
                           fmaxf(fmaxf(lg[4], lg[5]), fmaxf(lg[6], lg[7])));
    float sl = 0.0f;
#pragma unroll
    for (int j = 0; j < 8; ++j) sl += __expf(lg[j] - ml);
    const float nm = fmaxf(mx, ml);
    sm = sm * __expf(mx - nm) + sl * __expf(ml - nm);
    mx = nm;
  }

  red_m[wave][lane] = mx;
  red_s[wave][lane] = sm;
  __syncthreads();

  if (wave == 0 && lane < 16) {
    float M = -3.0e38f, S = 0.0f;
#pragma unroll
    for (int w = 0; w < 4; ++w) {
#pragma unroll
      for (int hh = 0; hh < 2; ++hh) {
        const float m2 = red_m[w][lane + 16 * hh];
        const float s2 = red_s[w][lane + 16 * hh];
        const float Mn = fmaxf(M, m2);
        S = S * __expf(M - Mn) + s2 * __expf(m2 - Mn);
        M = Mn;
      }
    }
    Mcol[(long)b * L_ + k0 + lane] = M;
    Scol[(long)b * L_ + k0 + lane] = S;
  }
}

// ---- Pass 2: att = P @ X with score recompute, fused normalize -----------
__global__ void k_attn_out(const bf16* __restrict__ xb, const bf16* __restrict__ xt,
                           const unsigned char* __restrict__ mask,
                           const float* __restrict__ Mcol, const float* __restrict__ Scol,
                           float* __restrict__ out) {
  __shared__ __attribute__((aligned(16))) bf16 Xq[16][XPAD];  // q-tile, reused 32x
  __shared__ __attribute__((aligned(16))) bf16 Psh[16][72];   // 16 q x 64 k (+pad)
  const int qt   = blockIdx.x % (L_ / 16);
  const int b    = blockIdx.x / (L_ / 16);
  const int q0   = qt * 16;
  const int wave = threadIdx.x >> 5;
  const int lane = threadIdx.x & 31;
  const int n    = lane & 15;
  const int half = lane >> 4;
  const int dw   = wave * 128;  // this wave's 128-wide d chunk

  const v8f zero = {};
  v8f acc[8];
#pragma unroll
  for (int t = 0; t < 8; ++t) acc[t] = zero;

  const bf16* xbase = xb + (long)b * L_ * H_;

  // stage rows q0..q0+15 into LDS (async); reused by every k-chunk below
  stage_tile_async(xbase + (long)q0 * H_, Xq, threadIdx.x);
  __syncthreads();

  for (int k0 = 0; k0 < L_; k0 += 64) {
    // ---- phase A: this wave computes S tile for k-subtile ks..ks+15 ----
    const int ks = k0 + 16 * wave;
    const bf16* bRow = xbase + (long)(ks + n) * H_;
    const float mc    = Mcol[(long)b * L_ + ks + n];
    const float inv_s = 1.0f / Scol[(long)b * L_ + ks + n];
    const unsigned char* mrow = mask + ((long)b * L_ + q0) * L_ + (ks + n);
    unsigned char mv[8];
#pragma unroll
    for (int j = 0; j < 8; ++j) mv[j] = mrow[(long)(j + 8 * half) * L_];

    v8f s = {};
#pragma unroll 4
    for (int d0 = 0; d0 < H_; d0 += 32) {
      v16bf a  = load_opA(&Xq[n][d0], half);   // ds_load_b128 x2
      v16bf bm = load_opB(bRow + d0, half);
      s = __builtin_amdgcn_wmma_f32_16x16x32_bf16(false, a, false, bm,
                                                  (short)0, s, false, false);
    }
#pragma unroll
    for (int j = 0; j < 8; ++j) {
      const int   mr    = j + 8 * half;
      const float logit = (mv[j] ? -1.0e9f : s[j]) * INV_SQRT_DK;
      const float p     = __expf(logit - mc) * inv_s;
      Psh[mr][16 * wave + n] = (bf16)p;
    }
    __syncthreads();

    // ---- phase B: acc[q, d] += P(16x32) @ X(32x16) over the 64-k chunk --
#pragma unroll
    for (int ks2 = 0; ks2 < 64; ks2 += 32) {
      v16bf a = load_opA(&Psh[n][ks2], half);  // m = lane&15 == n
#pragma unroll
      for (int t = 0; t < 8; ++t) {
        const bf16* krun = xt + ((long)b * H_ + dw + t * 16 + n) * L_ + k0 + ks2;
        v16bf bm = load_opB(krun, half);
        acc[t] = __builtin_amdgcn_wmma_f32_16x16x32_bf16(false, a, false, bm,
                                                         (short)0, acc[t], false, false);
      }
    }
    __syncthreads();
  }

  // ---- store f32 output: lane holds rows m = j + 8*half, col n ----------
#pragma unroll
  for (int t = 0; t < 8; ++t) {
#pragma unroll
    for (int j = 0; j < 8; ++j) {
      const int mr = j + 8 * half;
      out[((long)b * L_ + q0 + mr) * H_ + dw + t * 16 + n] = acc[t][j];
    }
  }
}

// --------------------------------------------------------------------------
extern "C" void kernel_launch(void* const* d_in, const int* in_sizes, int n_in,
                              void* d_out, int out_size, void* d_ws, size_t ws_size,
                              hipStream_t stream) {
  const float*         x    = (const float*)d_in[0];
  const unsigned char* mask = (const unsigned char*)d_in[1];  // jax bool -> u8
  float*               out  = (float*)d_out;

  char* ws = (char*)d_ws;
  bf16*  xb   = (bf16*)ws;                                    // 16 MB
  bf16*  xt   = (bf16*)(ws + (size_t)B_ * L_ * H_ * 2);       // 16 MB
  float* Mcol = (float*)(ws + (size_t)B_ * L_ * H_ * 4);      // 64 KB
  float* Scol = Mcol + (size_t)B_ * L_;                       // 64 KB

  k_convert<<<B_ * (L_ / 64) * (H_ / 64), 256, 0, stream>>>(x, xb, xt);
  k_colstats<<<B_ * (L_ / 16), 128, 0, stream>>>(xb, mask, Mcol, Scol);
  k_attn_out<<<B_ * (L_ / 16), 128, 0, stream>>>(xb, xt, mask, Mcol, Scol, out);
}
